// LATTICEModel_41996190220481
// MI455X (gfx1250) — matrix operand order repaired and checked by reading, not compiled, not run on hardware
//
#include <hip/hip_runtime.h>
#include <hip/hip_bf16.h>
#include <math.h>

#define NUM_USERS 16384
#define NUM_ITEMS 8192
#define NALL      (NUM_USERS + NUM_ITEMS)
#define TOPK      10
#define L_M       0.7f
#define NUM_EDGES 1048576

typedef __attribute__((ext_vector_type(16))) _Float16 v16h;
typedef __attribute__((ext_vector_type(8)))  _Float16 v8h;
typedef __attribute__((ext_vector_type(8)))  float    v8f;

// ---------------------------------------------------------------------------
// Row L2-normalize (f32 -> f16), block = 256 per row, generic K
// ---------------------------------------------------------------------------
__global__ __launch_bounds__(256)
void k_rownorm_f16(const float* __restrict__ src, _Float16* __restrict__ dst, int K) {
    __shared__ float red[256];
    const int row = blockIdx.x;
    const float* s = src + (size_t)row * K;
    float acc = 0.f;
    for (int i = threadIdx.x; i < K; i += 256) { float v = s[i]; acc += v * v; }
    red[threadIdx.x] = acc;
    __syncthreads();
    for (int off = 128; off > 0; off >>= 1) {
        if (threadIdx.x < off) red[threadIdx.x] += red[threadIdx.x + off];
        __syncthreads();
    }
    const float inv = 1.0f / fmaxf(sqrtf(red[0]), 1e-12f);
    for (int i = threadIdx.x; i < K; i += 256)
        dst[(size_t)row * K + i] = (_Float16)(s[i] * inv);
}

// Row L2-normalize for 64-wide rows (f32 -> f32, in-place safe)
__global__ __launch_bounds__(64)
void k_rownorm64(const float* __restrict__ src, float* __restrict__ dst) {
    __shared__ float red[64];
    const int r = blockIdx.x, d = threadIdx.x;
    const float v = src[(size_t)r * 64 + d];
    red[d] = v * v;
    __syncthreads();
    for (int off = 32; off > 0; off >>= 1) { if (d < off) red[d] += red[d + off]; __syncthreads(); }
    dst[(size_t)r * 64 + d] = v / fmaxf(sqrtf(red[0]), 1e-12f);
}

// W[K x 64] f32 -> Wt[64 x K] f16 (for WMMA B fragments)
__global__ void k_transposeW(const float* __restrict__ W, _Float16* __restrict__ Wt, int K) {
    const int i = blockIdx.x * 256 + threadIdx.x;
    if (i < K * 64) {
        const int k = i >> 6, n = i & 63;
        Wt[(size_t)n * K + k] = (_Float16)W[(size_t)k * 64 + n];
    }
}

// ---------------------------------------------------------------------------
// proj = fn @ W + b via WMMA.  grid = 512 blocks x 128 thr (4 waves).
// Wave w computes output columns [16w, 16w+16).
// ---------------------------------------------------------------------------
__global__ __launch_bounds__(128)
void k_proj_wmma(const _Float16* __restrict__ fn, const _Float16* __restrict__ Wt,
                 const float* __restrict__ b, float* __restrict__ proj, int K) {
    __shared__ _Float16 Alds[16 * 768];
    const int m0    = blockIdx.x * 16;
    const int tid   = threadIdx.x;
    const int wave  = tid >> 5;
    const int lane  = tid & 31;
    const int row16 = lane & 15;
    const int hi    = lane >> 4;

    for (int idx = tid; idx < 16 * (K / 8); idx += 128) {
        const int r  = idx / (K / 8);
        const int kk = (idx - r * (K / 8)) * 8;
        *(v8h*)&Alds[r * K + kk] = *(const v8h*)&fn[(size_t)(m0 + r) * K + kk];
    }
    __syncthreads();

    const int n0 = wave * 16;
    v8f acc = {};
    for (int k0 = 0; k0 < K; k0 += 32) {
        const _Float16* ap = &Alds[row16 * K + k0 + hi * 16];
        v8h alo = *(const v8h*)ap;
        v8h ahi = *(const v8h*)(ap + 8);
        v16h a = __builtin_shufflevector(alo, ahi,
                  0,1,2,3,4,5,6,7,8,9,10,11,12,13,14,15);
        v16h bf = *(const v16h*)&Wt[(size_t)(n0 + row16) * K + k0 + hi * 16];
        acc = __builtin_amdgcn_wmma_f32_16x16x32_f16(
                  false, a, false, bf, (short)0, acc, false, false);
    }
    const float bias = b[n0 + row16];
#pragma unroll
    for (int r = 0; r < 8; ++r)
        proj[(size_t)(m0 + hi * 8 + r) * 64 + n0 + row16] = acc[r] + bias;
}

// ---------------------------------------------------------------------------
// Fused  sim = P @ P^T  +  row-wise top-10   (P: [8192 x K] f16, row-major)
// Block = one 16-row M tile, 8 waves; each wave: 2 N-tiles/iter (shared A frag,
// double accumulators -> 2 WMMAs per A fragment load).
// ---------------------------------------------------------------------------
__global__ __launch_bounds__(256)
void k_sim_topk(const _Float16* __restrict__ P, int K,
                int* __restrict__ knn_cols, float* __restrict__ knn_vals) {
    __shared__ _Float16 Alds[16 * 768];          // max K = 768
    __shared__ float    simC[8][2][16][17];      // per-wave double-buffered C
    __shared__ float    topv[8][16][TOPK];
    __shared__ int      topi[8][16][TOPK];

    const int m0    = blockIdx.x * 16;
    const int tid   = threadIdx.x;
    const int wave  = tid >> 5;
    const int lane  = tid & 31;
    const int row16 = lane & 15;
    const int hi    = lane >> 4;

    for (int idx = tid; idx < 16 * (K / 8); idx += 256) {
        const int r  = idx / (K / 8);
        const int kk = (idx - r * (K / 8)) * 8;
        *(v8h*)&Alds[r * K + kk] = *(const v8h*)&P[(size_t)(m0 + r) * K + kk];
    }
    for (int idx = tid; idx < 8 * 16 * TOPK; idx += 256) {
        (&topv[0][0][0])[idx] = -1e30f;
        (&topi[0][0][0])[idx] = 0;
    }
    __syncthreads();

    const int NITER = (NUM_ITEMS / 16) / 16;     // 32
    for (int it = 0; it < NITER; ++it) {
        const int nt = it * 16 + wave * 2;       // this wave's tile pair
        const int n0 = nt * 16;
        v8f acc0 = {}, acc1 = {};
        if (it + 1 < NITER)
            __builtin_prefetch(&P[(size_t)((nt + 16) * 16 + row16) * K], 0, 1);

        for (int k0 = 0; k0 < K; k0 += 32) {
            const _Float16* ap = &Alds[row16 * K + k0 + hi * 16];
            v8h alo = *(const v8h*)ap;
            v8h ahi = *(const v8h*)(ap + 8);
            v16h a = __builtin_shufflevector(alo, ahi,
                      0,1,2,3,4,5,6,7,8,9,10,11,12,13,14,15);
            v16h b0 = *(const v16h*)&P[(size_t)(n0 + row16) * K + k0 + hi * 16];
            v16h b1 = *(const v16h*)&P[(size_t)(n0 + 16 + row16) * K + k0 + hi * 16];
            acc0 = __builtin_amdgcn_wmma_f32_16x16x32_f16(
                       false, a, false, b0, (short)0, acc0, false, false);
            acc1 = __builtin_amdgcn_wmma_f32_16x16x32_f16(
                       false, a, false, b1, (short)0, acc1, false, false);
        }
#pragma unroll
        for (int r = 0; r < 8; ++r) {
            simC[wave][0][hi * 8 + r][row16] = acc0[r];
            simC[wave][1][hi * 8 + r][row16] = acc1[r];
        }
        __syncthreads();
        if (hi == 0) {
            float* tv = topv[wave][row16];
            int*   ti = topi[wave][row16];
            for (int bq = 0; bq < 2; ++bq)
                for (int j = 0; j < 16; ++j) {
                    const float v = simC[wave][bq][row16][j];
                    int   mi = 0; float mv = tv[0];
#pragma unroll
                    for (int q = 1; q < TOPK; ++q) if (tv[q] < mv) { mv = tv[q]; mi = q; }
                    if (v > mv) { tv[mi] = v; ti[mi] = n0 + bq * 16 + j; }
                }
        }
        __syncthreads();
    }

    // Merge 8 waves' candidates per row -> global top-10 (descending)
    if (tid < 16) {
        const int row = tid;
        float bestv[TOPK]; int besti[TOPK];
        for (int t = 0; t < TOPK; ++t) {
            float bv = -1e30f; int bw = 0, bq = 0;
            for (int w = 0; w < 8; ++w)
                for (int q = 0; q < TOPK; ++q) {
                    const float v = topv[w][row][q];
                    if (v > bv) { bv = v; bw = w; bq = q; }
                }
            bestv[t] = bv; besti[t] = topi[bw][row][bq];
            topv[bw][row][bq] = -2e30f;
        }
        const int grow = m0 + row;
        for (int t = 0; t < TOPK; ++t) {
            knn_cols[grow * TOPK + t] = besti[t];
            knn_vals[grow * TOPK + t] = bestv[t];
        }
    }
}

// ---------------------------------------------------------------------------
// norm_adj machinery on slot-structured COO
// ---------------------------------------------------------------------------
__global__ void k_deg_knn(const int* __restrict__ cols, float* __restrict__ deg) {
    const int r = blockIdx.x * 256 + threadIdx.x;
    if (r >= NUM_ITEMS) return;
    int c[TOPK];
    for (int j = 0; j < TOPK; ++j) c[j] = cols[r * TOPK + j];
    int cnt = 0;
    for (int j = 0; j < TOPK; ++j) {
        bool dup = false;
        for (int q = 0; q < j; ++q) if (c[q] == c[j]) { dup = true; break; }
        if (!dup) ++cnt;
    }
    deg[r] = (float)cnt;
}

__global__ void k_level1(const int* __restrict__ kc, const float* __restrict__ kv,
                         const float* __restrict__ deg, const float* __restrict__ w_imp,
                         int m, int* __restrict__ c1, float* __restrict__ v1) {
    const int r = blockIdx.x * 256 + threadIdx.x;
    if (r >= NUM_ITEMS) return;
    const float e0 = expf(w_imp[0]), e1 = expf(w_imp[1]);
    const float w = (m == 0 ? e0 : e1) / (e0 + e1);
    const float dr = deg[r];
    const float dinvr = dr > 0.f ? rsqrtf(dr) : 0.f;
    for (int j = 0; j < TOPK; ++j) {
        const int c = kc[r * TOPK + j];
        float v = kv[r * TOPK + j];
        if (c == r) v = 0.f;
        const float dc = deg[c];
        const float dinvc = dc > 0.f ? rsqrtf(dc) : 0.f;
        c1[r * 11 + j] = c;
        v1[r * 11 + j] = v * dinvr * dinvc * w;
    }
    c1[r * 11 + TOPK] = r;
    v1[r * 11 + TOPK] = dinvr * dinvr * w;
}

__global__ void k_deg22(const int* __restrict__ cA, const int* __restrict__ cB,
                        float* __restrict__ deg) {
    const int r = blockIdx.x * 256 + threadIdx.x;
    if (r >= NUM_ITEMS) return;
    int c[22];
    for (int j = 0; j < 11; ++j) { c[j] = cA[r * 11 + j]; c[11 + j] = cB[r * 11 + j]; }
    int cnt = 0;
    for (int j = 0; j < 22; ++j) {
        bool dup = false;
        for (int q = 0; q < j; ++q) if (c[q] == c[j]) { dup = true; break; }
        if (!dup) ++cnt;
    }
    deg[r] = (float)cnt;
}

__global__ void k_level2(const int* __restrict__ cA, const float* __restrict__ vA,
                         const int* __restrict__ cB, const float* __restrict__ vB,
                         const float* __restrict__ deg,
                         int* __restrict__ c2, float* __restrict__ v2) {
    const int r = blockIdx.x * 256 + threadIdx.x;
    if (r >= NUM_ITEMS) return;
    const float dr = deg[r];
    const float dinvr = dr > 0.f ? rsqrtf(dr) : 0.f;
    for (int j = 0; j < 22; ++j) {
        int c; float v;
        if (j < 11) { c = cA[r * 11 + j]; v = vA[r * 11 + j]; }
        else        { c = cB[r * 11 + (j - 11)]; v = vB[r * 11 + (j - 11)]; }
        if (c == r) v = 0.f;
        const float dc = deg[c];
        const float dinvc = dc > 0.f ? rsqrtf(dc) : 0.f;
        c2[r * 23 + j] = c;
        v2[r * 23 + j] = v * dinvr * dinvc;
    }
    c2[r * 23 + 22] = r;
    v2[r * 23 + 22] = dinvr * dinvr;
}

__global__ __launch_bounds__(64)
void k_spmm_item(const int* __restrict__ cL, const float* __restrict__ vL,
                 const int* __restrict__ cO, const float* __restrict__ vO,
                 const float* __restrict__ Gi, float* __restrict__ x) {
    const int r = blockIdx.x, d = threadIdx.x;
    float acc = 0.f;
    for (int j = 0; j < 23; ++j) {
        acc += (1.0f - L_M) * vL[r * 23 + j] * Gi[(size_t)cL[r * 23 + j] * 64 + d];
        acc += L_M          * vO[r * 23 + j] * Gi[(size_t)cO[r * 23 + j] * 64 + d];
    }
    x[(size_t)r * 64 + d] = acc;
}

// ---------------------------------------------------------------------------
// User-item graph: deterministic CSR build + gather SpMM (no f32 atomics)
// ---------------------------------------------------------------------------
__global__ void k_zero_int(int* __restrict__ p, int n) {
    const int i = blockIdx.x * 256 + threadIdx.x;
    if (i < n) p[i] = 0;
}
__global__ void k_copy_int(const int* __restrict__ s, int* __restrict__ d, int n) {
    const int i = blockIdx.x * 256 + threadIdx.x;
    if (i < n) d[i] = s[i];
}
__global__ void k_count(const int* __restrict__ rows, int* __restrict__ cnt) {
    const int e = blockIdx.x * 256 + threadIdx.x;
    if (e < NUM_EDGES) atomicAdd(&cnt[rows[e]], 1);
}
// Exclusive scan of cnt[0..NALL) -> rowptr[0..NALL], one block of 1024 threads,
// 24 elements per thread (24576 = 1024*24).
__global__ __launch_bounds__(1024)
void k_scan(const int* __restrict__ cnt, int* __restrict__ rowptr) {
    __shared__ int s[1024];
    const int t = threadIdx.x;
    const int base = t * 24;
    int loc[24];
    int sum = 0;
#pragma unroll
    for (int j = 0; j < 24; ++j) { loc[j] = sum; sum += cnt[base + j]; }
    s[t] = sum;
    __syncthreads();
    for (int off = 1; off < 1024; off <<= 1) {
        const int v = (t >= off) ? s[t - off] : 0;
        __syncthreads();
        s[t] += v;
        __syncthreads();
    }
    const int pre = (t == 0) ? 0 : s[t - 1];
#pragma unroll
    for (int j = 0; j < 24; ++j) rowptr[base + j] = pre + loc[j];
    if (t == 1023) rowptr[NALL] = pre + sum;
}
__global__ void k_fill(const int* __restrict__ rows, int* __restrict__ cur,
                       int* __restrict__ eidx) {
    const int e = blockIdx.x * 256 + threadIdx.x;
    if (e < NUM_EDGES) { const int p = atomicAdd(&cur[rows[e]], 1); eidx[p] = e; }
}
// Sort each row's edge ids ascending -> deterministic summation order
__global__ void k_sort_rows(const int* __restrict__ rowptr, int* __restrict__ eidx) {
    const int r = blockIdx.x * 256 + threadIdx.x;
    if (r >= NALL) return;
    const int s = rowptr[r], e = rowptr[r + 1];
    for (int i = s + 1; i < e; ++i) {
        const int key = eidx[i];
        int j = i - 1;
        while (j >= s && eidx[j] > key) { eidx[j + 1] = eidx[j]; --j; }
        eidx[j + 1] = key;
    }
}
__global__ __launch_bounds__(64)
void k_spmm_csr(const int* __restrict__ rowptr, const int* __restrict__ eidx,
                const int* __restrict__ cols, const float* __restrict__ vals,
                const float* __restrict__ xin, float* __restrict__ h) {
    const int r = blockIdx.x, d = threadIdx.x;
    const int s = rowptr[r], e = rowptr[r + 1];
    float acc = 0.f;
    for (int p = s; p < e; ++p) {
        const int ed = eidx[p];
        acc += vals[ed] * xin[(size_t)cols[ed] * 64 + d];
    }
    h[(size_t)r * 64 + d] = acc;
}

__global__ void k_copy_ego(const float* __restrict__ Gu, const float* __restrict__ Gi,
                           float* __restrict__ x0) {
    const int i = blockIdx.x * 256 + threadIdx.x;
    const int NU = NUM_USERS * 64;
    if (i < NU) x0[i] = Gu[i];
    else if (i < NALL * 64) x0[i] = Gi[i - NU];
}

__global__ __launch_bounds__(64)
void k_final(const float* __restrict__ x0, const float* __restrict__ x1,
             const float* __restrict__ x2, const float* __restrict__ xi,
             float* __restrict__ out) {
    __shared__ float red[64];
    const int r = blockIdx.x, d = threadIdx.x;
    float a = (x0[(size_t)r * 64 + d] + x1[(size_t)r * 64 + d] + x2[(size_t)r * 64 + d]) * (1.f / 3.f);
    if (r >= NUM_USERS) {
        const int ir = r - NUM_USERS;
        const float v = xi[(size_t)ir * 64 + d];
        red[d] = v * v;
        __syncthreads();
        for (int off = 32; off > 0; off >>= 1) { if (d < off) red[d] += red[d + off]; __syncthreads(); }
        a += v / fmaxf(sqrtf(red[0]), 1e-12f);
    }
    out[(size_t)r * 64 + d] = a;
}

// ---------------------------------------------------------------------------
// Orchestration
// ---------------------------------------------------------------------------
extern "C" void kernel_launch(void* const* d_in, const int* in_sizes, int n_in,
                              void* d_out, int out_size, void* d_ws, size_t ws_size,
                              hipStream_t stream) {
    (void)in_sizes; (void)n_in; (void)out_size; (void)ws_size;
    const float* feats_v  = (const float*)d_in[0];
    const float* feats_t  = (const float*)d_in[1];
    const float* W_v      = (const float*)d_in[2];
    const float* b_v      = (const float*)d_in[3];
    const float* W_t      = (const float*)d_in[4];
    const float* b_t      = (const float*)d_in[5];
    const float* w_imp    = (const float*)d_in[6];
    const float* Gu       = (const float*)d_in[7];
    const float* Gi       = (const float*)d_in[8];
    const float* adj_vals = (const float*)d_in[9];
    const int*   adj_rows = (const int*)d_in[10];
    const int*   adj_cols = (const int*)d_in[11];
    float* out = (float*)d_out;

    char* ws = (char*)d_ws;
    size_t off = 0;
    auto alloc = [&](size_t bytes) -> void* {
        void* p = ws + off;
        off += (bytes + 255) & ~(size_t)255;
        return p;
    };

    _Float16* fn[2];
    fn[0] = (_Float16*)alloc((size_t)NUM_ITEMS * 768 * 2);
    fn[1] = (_Float16*)alloc((size_t)NUM_ITEMS * 384 * 2);
    _Float16* pn[2];
    pn[0] = (_Float16*)alloc((size_t)NUM_ITEMS * 64 * 2);
    pn[1] = (_Float16*)alloc((size_t)NUM_ITEMS * 64 * 2);
    _Float16* Wt = (_Float16*)alloc((size_t)64 * 768 * 2);
    float* proj  = (float*)alloc((size_t)NUM_ITEMS * 64 * 4);
    int*   knn_c = (int*)alloc((size_t)NUM_ITEMS * TOPK * 4);
    float* knn_v = (float*)alloc((size_t)NUM_ITEMS * TOPK * 4);
    float* deg   = (float*)alloc((size_t)NUM_ITEMS * 4);
    int* cLm[2]; float* vLm[2]; int* cOm[2]; float* vOm[2];
    for (int m = 0; m < 2; ++m) {
        cLm[m] = (int*)alloc((size_t)NUM_ITEMS * 11 * 4);
        vLm[m] = (float*)alloc((size_t)NUM_ITEMS * 11 * 4);
        cOm[m] = (int*)alloc((size_t)NUM_ITEMS * 11 * 4);
        vOm[m] = (float*)alloc((size_t)NUM_ITEMS * 11 * 4);
    }
    int*   c2L = (int*)alloc((size_t)NUM_ITEMS * 23 * 4);
    float* v2L = (float*)alloc((size_t)NUM_ITEMS * 23 * 4);
    int*   c2O = (int*)alloc((size_t)NUM_ITEMS * 23 * 4);
    float* v2O = (float*)alloc((size_t)NUM_ITEMS * 23 * 4);
    float* xitem = (float*)alloc((size_t)NUM_ITEMS * 64 * 4);
    float* x0 = (float*)alloc((size_t)NALL * 64 * 4);
    float* x1 = (float*)alloc((size_t)NALL * 64 * 4);
    float* x2 = (float*)alloc((size_t)NALL * 64 * 4);
    int* cnt    = (int*)alloc((size_t)NALL * 4);
    int* rowptr = (int*)alloc((size_t)(NALL + 1) * 4);
    int* cur    = (int*)alloc((size_t)NALL * 4);
    int* eidx   = (int*)alloc((size_t)NUM_EDGES * 4);

    const float* feats[2] = {feats_v, feats_t};
    const float* Wm[2]    = {W_v, W_t};
    const float* bm[2]    = {b_v, b_t};
    const int    Kf[2]    = {768, 384};

    for (int m = 0; m < 2; ++m) {
        const int K = Kf[m];
        k_rownorm_f16<<<NUM_ITEMS, 256, 0, stream>>>(feats[m], fn[m], K);
        k_transposeW<<<(K * 64 + 255) / 256, 256, 0, stream>>>(Wm[m], Wt, K);
        k_proj_wmma<<<NUM_ITEMS / 16, 128, 0, stream>>>(fn[m], Wt, bm[m], proj, K);
        k_rownorm_f16<<<NUM_ITEMS, 256, 0, stream>>>(proj, pn[m], 64);
        // learned graph: knn on l2norm(proj)
        k_sim_topk<<<NUM_ITEMS / 16, 256, 0, stream>>>(pn[m], 64, knn_c, knn_v);
        k_deg_knn<<<NUM_ITEMS / 256, 256, 0, stream>>>(knn_c, deg);
        k_level1<<<NUM_ITEMS / 256, 256, 0, stream>>>(knn_c, knn_v, deg, w_imp, m, cLm[m], vLm[m]);
        // original graph: knn on l2norm(feats)
        k_sim_topk<<<NUM_ITEMS / 16, 256, 0, stream>>>(fn[m], K, knn_c, knn_v);
        k_deg_knn<<<NUM_ITEMS / 256, 256, 0, stream>>>(knn_c, deg);
        k_level1<<<NUM_ITEMS / 256, 256, 0, stream>>>(knn_c, knn_v, deg, w_imp, m, cOm[m], vOm[m]);
    }
    // second-level normalization of concatenated graphs
    k_deg22<<<NUM_ITEMS / 256, 256, 0, stream>>>(cLm[0], cLm[1], deg);
    k_level2<<<NUM_ITEMS / 256, 256, 0, stream>>>(cLm[0], vLm[0], cLm[1], vLm[1], deg, c2L, v2L);
    k_deg22<<<NUM_ITEMS / 256, 256, 0, stream>>>(cOm[0], cOm[1], deg);
    k_level2<<<NUM_ITEMS / 256, 256, 0, stream>>>(cOm[0], vOm[0], cOm[1], vOm[1], deg, c2O, v2O);
    // item-item propagation (N_LAYERS = 1)
    k_spmm_item<<<NUM_ITEMS, 64, 0, stream>>>(c2L, v2L, c2O, v2O, Gi, xitem);

    // deterministic CSR of the user-item graph
    k_zero_int<<<(NALL + 255) / 256, 256, 0, stream>>>(cnt, NALL);
    k_count<<<(NUM_EDGES + 255) / 256, 256, 0, stream>>>(adj_rows, cnt);
    k_scan<<<1, 1024, 0, stream>>>(cnt, rowptr);
    k_copy_int<<<(NALL + 255) / 256, 256, 0, stream>>>(rowptr, cur, NALL);
    k_fill<<<(NUM_EDGES + 255) / 256, 256, 0, stream>>>(adj_rows, cur, eidx);
    k_sort_rows<<<(NALL + 255) / 256, 256, 0, stream>>>(rowptr, eidx);

    // user-item propagation (N_UI_LAYERS = 2)
    const int nall_e = NALL * 64;
    k_copy_ego<<<(nall_e + 255) / 256, 256, 0, stream>>>(Gu, Gi, x0);
    k_spmm_csr<<<NALL, 64, 0, stream>>>(rowptr, eidx, adj_cols, adj_vals, x0, x1);
    k_rownorm64<<<NALL, 64, 0, stream>>>(x1, x1);
    k_spmm_csr<<<NALL, 64, 0, stream>>>(rowptr, eidx, adj_cols, adj_vals, x1, x2);
    k_rownorm64<<<NALL, 64, 0, stream>>>(x2, x2);

    k_final<<<NALL, 64, 0, stream>>>(x0, x1, x2, xitem, out);
}
